// DeFTAN2block_10247791968687
// MI455X (gfx1250) — matrix operand order repaired and checked by reading, not compiled, not run on hardware
//
#include <hip/hip_runtime.h>
#include <hip/hip_bf16.h>
#include <math.h>

// ---------------------------------------------------------------------------
// DeFTAN-2 block for gfx1250 (MI455X).  All GEMM-shaped work (q/k/v/out
// projections, attention einsums, FFN pw1/pw2/pw3) runs on
// v_wmma_f32_16x16x32_f16 with f32 accumulation; convs / norms / softmaxes
// stay f32 VALU.  Fragment loads are branchless (vectorized b128 where the
// source is contiguous, address-clamped + select-zero where guarded).
// GEMMs compute a 16x64 tile per wave (1 A fragment x 4 B fragments -> 4 WMMA
// per K-step) and are templated on bias/residual/activation so epilogues are
// branch-free.
// ---------------------------------------------------------------------------

typedef __attribute__((ext_vector_type(16))) _Float16 v16h;
typedef __attribute__((ext_vector_type(8)))  float    v8f;
typedef __attribute__((ext_vector_type(4)))  float    v4f;

#define ACT_NONE 0
#define ACT_GELU 1

__device__ __forceinline__ int laneid() { return threadIdx.x & 31; }

// ---- WMMA fragment loaders (f32 source -> f16 fragment), contiguous-k order, same in A and B ----
// A fragment, 16(M) x 32(K) f16.  lanes 0-15: M=lane, K = k0+0..15
//                                 lanes16-31: M=lane-16, K = k0+16..31

// Fast path: no guards (M%16==0 slice fully valid, K%32==0).  4x b128 loads.
__device__ __forceinline__ v16h frag_a_rm_fast(const float* __restrict__ src, long ld,
                                               int m0, int k0) {
  int l = laneid();
  int m = m0 + (l & 15);
  int kb = k0 + ((l & 16) ? 16 : 0);
  const float* p = src + (long)m * ld + kb;
  v4f a0 = *(const v4f*)(p);
  v4f a1 = *(const v4f*)(p + 4);
  v4f a2 = *(const v4f*)(p + 8);
  v4f a3 = *(const v4f*)(p + 12);
  v16h f;
#pragma unroll
  for (int i = 0; i < 4; ++i) {
    f[i]      = (_Float16)a0[i];
    f[i + 4]  = (_Float16)a1[i];
    f[i + 8]  = (_Float16)a2[i];
    f[i + 12] = (_Float16)a3[i];
  }
  return f;
}

// Row-clamped variant: K fully valid, row may exceed mLim (clamped address,
// whole-fragment select to zero; per-lane uniform -> still vectorized loads).
__device__ __forceinline__ v16h frag_a_rm_rowclamp(const float* __restrict__ src, long ld,
                                                   int m0, int mLim, int k0) {
  int l = laneid();
  int m = m0 + (l & 15);
  bool ok = (m < mLim);
  int mc = ok ? m : (mLim - 1);
  int kb = k0 + ((l & 16) ? 16 : 0);
  const float* p = src + (long)mc * ld + kb;
  v4f a0 = *(const v4f*)(p);
  v4f a1 = *(const v4f*)(p + 4);
  v4f a2 = *(const v4f*)(p + 8);
  v4f a3 = *(const v4f*)(p + 12);
  _Float16 zero = (_Float16)0.0f;
  v16h f;
#pragma unroll
  for (int i = 0; i < 4; ++i) {
    f[i]      = ok ? (_Float16)a0[i] : zero;
    f[i + 4]  = ok ? (_Float16)a1[i] : zero;
    f[i + 8]  = ok ? (_Float16)a2[i] : zero;
    f[i + 12] = ok ? (_Float16)a3[i] : zero;
  }
  return f;
}

// A fragment from transposed view A[m,k] = src[k*ld + m], K guarded
// (branchless: clamp k address, select zero).
__device__ __forceinline__ v16h frag_a_tr_kclamp(const float* __restrict__ src, long ld,
                                                 int m0, int k0, int kLim) {
  int l = laneid();
  int m = m0 + (l & 15);
  int kb = k0 + ((l & 16) ? 16 : 0);
  _Float16 zero = (_Float16)0.0f;
  v16h f;
#pragma unroll
  for (int i = 0; i < 8; ++i) {
    int ka = kb + i, kc = kb + 8 + i;
    int kaC = (ka < kLim) ? ka : 0;
    int kcC = (kc < kLim) ? kc : 0;
    float va = src[(long)kaC * ld + m];
    float vc = src[(long)kcC * ld + m];
    f[i]     = (ka < kLim) ? (_Float16)va : zero;
    f[i + 8] = (kc < kLim) ? (_Float16)vc : zero;
  }
  return f;
}

// B fragment, 32(K) x 16(N) f16.  lanes 0-15: N=lane, K = k0+0..15;
//                                 lanes16-31: K = k0+16..31.
// Fast path from N x K row-major weight (no guards).  4x b128 loads.
__device__ __forceinline__ v16h frag_b_nt_fast(const float* __restrict__ w, long ld,
                                               int n0, int k0) {
  int l = laneid();
  int n = n0 + (l & 15);
  int kb = k0 + ((l & 16) ? 16 : 0);
  const float* p = w + (long)n * ld + kb;
  v4f b0 = *(const v4f*)(p);
  v4f b1 = *(const v4f*)(p + 4);
  v4f b2 = *(const v4f*)(p + 8);
  v4f b3 = *(const v4f*)(p + 12);
  v16h f;
#pragma unroll
  for (int i = 0; i < 4; ++i) {
    f[i]      = (_Float16)b0[i];
    f[i + 4]  = (_Float16)b1[i];
    f[i + 8]  = (_Float16)b2[i];
    f[i + 12] = (_Float16)b3[i];
  }
  return f;
}

// B fragment from K x N row-major matrix, no guards (strided b32 loads).
__device__ __forceinline__ v16h frag_b_nn_fast(const float* __restrict__ src, long ld,
                                               int n0, int k0) {
  int l = laneid();
  int n = n0 + (l & 15);
  int kb = k0 + ((l & 16) ? 16 : 0);
  v16h f;
#pragma unroll
  for (int i = 0; i < 16; ++i) f[i] = (_Float16)src[(long)(kb + i) * ld + n];
  return f;
}

// B fragment from K x N row-major matrix, K guarded (clamp + select zero).
__device__ __forceinline__ v16h frag_b_nn_kclamp(const float* __restrict__ src, long ld,
                                                 int n0, int k0, int kLim) {
  int l = laneid();
  int n = n0 + (l & 15);
  int kb = k0 + ((l & 16) ? 16 : 0);
  _Float16 zero = (_Float16)0.0f;
  v16h f;
#pragma unroll
  for (int i = 0; i < 16; ++i) {
    int k = kb + i;
    int kC = (k < kLim) ? k : 0;
    float v = src[(long)kC * ld + n];
    f[i] = (k < kLim) ? (_Float16)v : zero;
  }
  return f;
}

__device__ __forceinline__ v8f wmma_f16(v16h a, v16h b, v8f c) {
  return __builtin_amdgcn_wmma_f32_16x16x32_f16(false, a, false, b, (short)0, c, false, false);
}

__device__ __forceinline__ float gelu_exact(float x) {
  return 0.5f * x * (1.0f + erff(x * 0.70710678118654752f));
}

// ---------------------------------------------------------------------------
// 1) ln4d (normalize over C=64) + permute into (N, C, L) "feature" layout.
//    mode 0: n = b*T + t, L-index = q       mode 1: n = b*Q + q, L-index = t
// ---------------------------------------------------------------------------
__global__ void ln4d_permute_kernel(const float* __restrict__ x,
                                    const float* __restrict__ g,
                                    const float* __restrict__ bb,
                                    float* __restrict__ feat,
                                    int B, int T, int Q, int mode) {
  long id = (long)blockIdx.x * blockDim.x + threadIdx.x;
  long total = (long)B * T * Q;
  if (id >= total) return;
  int q = (int)(id % Q);
  long r = id / Q;
  int t = (int)(r % T);
  int b = (int)(r / T);
  long strideC = (long)T * Q;
  const float* p = x + ((long)b * 64) * strideC + (long)t * Q + q;
  float s = 0.f, ss = 0.f;
#pragma unroll 8
  for (int c = 0; c < 64; ++c) { float v = p[(long)c * strideC]; s += v; ss += v * v; }
  float mu = s * (1.0f / 64.0f);
  float var = ss * (1.0f / 64.0f) - mu * mu;
  float rinv = rsqrtf(var + 1e-5f);
  for (int c = 0; c < 64; ++c) {
    float v = (p[(long)c * strideC] - mu) * rinv * g[c] + bb[c];
    long dst = (mode == 0) ? (((long)(b * T + t) * 64 + c) * Q + q)
                           : (((long)(b * Q + q) * 64 + c) * T + t);
    feat[dst] = v;
  }
}

// ---------------------------------------------------------------------------
// 2) SDB stage: conv1d(k=3,pad=1) on "skip" -> per-sample global GN -> PReLU.
//    stage 0: in = feat[:, c, l]            (64 ch)
//    stage s: in = [prev(64ch), feat[:, c, l+s](64ch)]  (128 ch)
//    One workgroup (256 threads) per sample.
// ---------------------------------------------------------------------------
__global__ void sdb_stage_kernel(const float* __restrict__ feat, int Lin,
                                 const float* __restrict__ prev,
                                 const float* __restrict__ w,
                                 const float* __restrict__ cb,
                                 const float* __restrict__ gg,
                                 const float* __restrict__ gb,
                                 const float* __restrict__ pr,
                                 float* __restrict__ out, int Lu, int stage) {
  __shared__ float sS[256], sSS[256];
  int n = blockIdx.x, tid = threadIdx.x;
  int in_ch = (stage == 0) ? 64 : 128;
  const float* fbase = feat + (long)n * 64 * Lin;
  const float* pbase = prev ? (prev + (long)n * 64 * Lu) : nullptr;
  float* obase = out + (long)n * 64 * Lu;
  int tot = 64 * Lu;
  float s = 0.f, ss = 0.f;
  for (int idx = tid; idx < tot; idx += 256) {
    int co = idx / Lu, l = idx % Lu;
    float acc = cb[co];
    for (int dk = 0; dk < 3; ++dk) {
      int pos = l + dk - 1;
      if ((unsigned)pos >= (unsigned)Lu) continue;
      const float* wp = w + ((long)co * in_ch) * 3 + dk;
      if (stage == 0) {
        for (int ic = 0; ic < 64; ++ic) acc += fbase[(long)ic * Lin + pos] * wp[ic * 3];
      } else {
        for (int ic = 0; ic < 64; ++ic) acc += pbase[(long)ic * Lu + pos] * wp[ic * 3];
        for (int ic = 0; ic < 64; ++ic)
          acc += fbase[(long)ic * Lin + pos + stage] * wp[(64 + ic) * 3];
      }
    }
    obase[idx] = acc;
    s += acc; ss += acc * acc;
  }
  sS[tid] = s; sSS[tid] = ss; __syncthreads();
  for (int st = 128; st > 0; st >>= 1) {
    if (tid < st) { sS[tid] += sS[tid + st]; sSS[tid] += sSS[tid + st]; }
    __syncthreads();
  }
  float mu = sS[0] / (float)tot;
  float var = sSS[0] / (float)tot - mu * mu;
  float rinv = rsqrtf(var + 1e-5f);
  for (int idx = tid; idx < tot; idx += 256) {
    int co = idx / Lu;
    float v = (obase[idx] - mu) * rinv * gg[stage * 64 + co] + gb[stage * 64 + co];
    obase[idx] = (v >= 0.f) ? v : pr[stage * 64 + co] * v;
  }
}

// (N, C, Lu) -> token layout (N*Lu, 64)
__global__ void ncl_to_tok_kernel(const float* __restrict__ src, float* __restrict__ dst,
                                  int N, int Lu) {
  long id = (long)blockIdx.x * blockDim.x + threadIdx.x;
  long total = (long)N * 64 * Lu;
  if (id >= total) return;
  int l = (int)(id % Lu);
  long r = id / Lu;
  int c = (int)(r % 64);
  long n = r / 64;
  dst[((long)(n * Lu + l)) * 64 + c] = src[id];
}

// token layer-norm over 64 features
__global__ void token_ln_kernel(const float* __restrict__ y, const float* __restrict__ g,
                                const float* __restrict__ b, float* __restrict__ o, long Ntok) {
  long t = (long)blockIdx.x * blockDim.x + threadIdx.x;
  if (t >= Ntok) return;
  const float* p = y + t * 64;
  float s = 0.f, ss = 0.f;
#pragma unroll 8
  for (int c = 0; c < 64; ++c) { float v = p[c]; s += v; ss += v * v; }
  float mu = s * (1.0f / 64.0f);
  float var = ss * (1.0f / 64.0f) - mu * mu;
  float rinv = rsqrtf(var + 1e-5f);
  for (int c = 0; c < 64; ++c) o[t * 64 + c] = (p[c] - mu) * rinv * g[c] + b[c];
}

// ---------------------------------------------------------------------------
// 3) qk conv (k=3, pad=1, 64 -> 2*64) with GLU gating, on token layout.
// ---------------------------------------------------------------------------
__global__ void qkconv_glu_kernel(const float* __restrict__ yn, const float* __restrict__ w,
                                  float* __restrict__ qk, int N, int Lu) {
  long id = (long)blockIdx.x * blockDim.x + threadIdx.x;
  long total = (long)N * Lu * 64;
  if (id >= total) return;
  int oc = (int)(id % 64);
  long r = id / 64;
  int l = (int)(r % Lu);
  long n = r / Lu;
  const float* yb = yn + (long)n * Lu * 64;
  float a = 0.f, gt = 0.f;
  for (int dk = 0; dk < 3; ++dk) {
    int pos = l + dk - 1;
    if ((unsigned)pos >= (unsigned)Lu) continue;
    const float* yp = yb + (long)pos * 64;
    const float* wa = w + ((long)oc * 64) * 3 + dk;
    const float* wg = w + ((long)(oc + 64) * 64) * 3 + dk;
    for (int c = 0; c < 64; ++c) { a += yp[c] * wa[c * 3]; gt += yp[c] * wg[c * 3]; }
  }
  qk[id] = a * (1.0f / (1.0f + expf(-gt)));
}

// ---------------------------------------------------------------------------
// 4) Generic WMMA GEMM:  D[m, dcol0+n] = act( A[M,K] @ W[N,K]^T + bias + resid )
//    Requires M%16==0, N%64==0, K%32==0 (true for every call site).
//    One 16x64 tile per wave: 1 A fragment x 4 B fragments -> 4 WMMA per
//    K-step.  Templated epilogue -> branch-free.
// ---------------------------------------------------------------------------
template <int ACT, bool HAS_BIAS, bool HAS_RESID>
__global__ void gemm_nt_wmma_kernel(const float* __restrict__ A, const float* __restrict__ W,
                                    const float* __restrict__ bias,
                                    const float* __restrict__ resid,
                                    float* __restrict__ D, int M, int N, int K,
                                    int ldd, int dcol0) {
  int wid = blockIdx.x * (blockDim.x >> 5) + (threadIdx.x >> 5);
  int TN = N >> 6;
  int TM = M >> 4;
  if (wid >= TM * TN) return;
  int tm = wid / TN, tn = wid % TN;
  int m0 = tm << 4, n0 = tn << 6;
  v8f acc0 = {0.f, 0.f, 0.f, 0.f, 0.f, 0.f, 0.f, 0.f};
  v8f acc1 = acc0, acc2 = acc0, acc3 = acc0;
  for (int k0 = 0; k0 < K; k0 += 32) {
    v16h a = frag_a_rm_fast(A, K, m0, k0);
    acc0 = wmma_f16(a, frag_b_nt_fast(W, K, n0,      k0), acc0);
    acc1 = wmma_f16(a, frag_b_nt_fast(W, K, n0 + 16, k0), acc1);
    acc2 = wmma_f16(a, frag_b_nt_fast(W, K, n0 + 32, k0), acc2);
    acc3 = wmma_f16(a, frag_b_nt_fast(W, K, n0 + 48, k0), acc3);
  }
  int l = laneid();
  int cl = l & 15;
  int rbase = m0 + ((l & 16) ? 8 : 0);
  float b0 = 0.f, b1 = 0.f, b2 = 0.f, b3 = 0.f;
  if (HAS_BIAS) {
    b0 = bias[n0 + cl];
    b1 = bias[n0 + 16 + cl];
    b2 = bias[n0 + 32 + cl];
    b3 = bias[n0 + 48 + cl];
  }
#pragma unroll
  for (int r = 0; r < 8; ++r) {
    int row = rbase + r;
    float v0 = acc0[r] + b0;
    float v1 = acc1[r] + b1;
    float v2 = acc2[r] + b2;
    float v3 = acc3[r] + b3;
    if (HAS_RESID) {
      const float* rp = resid + (long)row * N + n0 + cl;
      v0 += rp[0];
      v1 += rp[16];
      v2 += rp[32];
      v3 += rp[48];
    }
    if (ACT == ACT_GELU) {
      v0 = gelu_exact(v0); v1 = gelu_exact(v1);
      v2 = gelu_exact(v2); v3 = gelu_exact(v3);
    }
    float* dp = D + (long)row * ldd + dcol0 + n0 + cl;
    dp[0]  = v0;
    dp[16] = v1;
    dp[32] = v2;
    dp[48] = v3;
  }
}

static void launch_gemm(const float* A, const float* W, const float* bias, const float* resid,
                        float* D, int M, int N, int K, int ldd, int dcol0, int act,
                        hipStream_t stream) {
  int tiles = (M >> 4) * (N >> 6);
  int blocks = (tiles + 7) / 8;
  if (act == ACT_GELU) {
    gemm_nt_wmma_kernel<ACT_GELU, true, false><<<blocks, 256, 0, stream>>>(
        A, W, bias, resid, D, M, N, K, ldd, dcol0);
  } else if (resid) {
    gemm_nt_wmma_kernel<ACT_NONE, true, true><<<blocks, 256, 0, stream>>>(
        A, W, bias, resid, D, M, N, K, ldd, dcol0);
  } else {
    gemm_nt_wmma_kernel<ACT_NONE, false, false><<<blocks, 256, 0, stream>>>(
        A, W, bias, resid, D, M, N, K, ldd, dcol0);
  }
}

// ---------------------------------------------------------------------------
// 5) Softmaxes.  k: over tokens per (sample, head*64+d) column (stride 256).
//                q: over 64 contiguous d per (token, head).
// ---------------------------------------------------------------------------
__global__ void softmax_col_kernel(float* __restrict__ k, int N, int Lu) {
  long id = (long)blockIdx.x * blockDim.x + threadIdx.x;
  if (id >= (long)N * 256) return;
  long n = id >> 8;
  int c = (int)(id & 255);
  float* p = k + n * (long)Lu * 256 + c;
  float mx = -1e30f;
  for (int l = 0; l < Lu; ++l) mx = fmaxf(mx, p[(long)l * 256]);
  float s = 0.f;
  for (int l = 0; l < Lu; ++l) s += expf(p[(long)l * 256] - mx);
  float inv = 1.0f / s;
  for (int l = 0; l < Lu; ++l) p[(long)l * 256] = expf(p[(long)l * 256] - mx) * inv;
}

__global__ void softmax_row64_kernel(float* __restrict__ q, long Ntok) {
  long id = (long)blockIdx.x * blockDim.x + threadIdx.x;
  if (id >= Ntok * 4) return;
  float* p = q + id * 64;
  float mx = -1e30f;
  for (int c = 0; c < 64; ++c) mx = fmaxf(mx, p[c]);
  float s = 0.f;
  for (int c = 0; c < 64; ++c) s += expf(p[c] - mx);
  float inv = 1.0f / s;
  for (int c = 0; c < 64; ++c) p[c] = expf(p[c] - mx) * inv;
}

// ---------------------------------------------------------------------------
// 6) kw[b,h,d,e] = SCALE * sum_n softk[n,d] * v[n,e]   (per sample-head, 64x64)
//    One block of 4 waves per (sample, head); wave w owns d-rows [16w,16w+16).
// ---------------------------------------------------------------------------
__global__ void attn_kw_wmma_kernel(const float* __restrict__ softk,
                                    const float* __restrict__ vbuf,
                                    float* __restrict__ kw, int Lu) {
  int nh = blockIdx.x;          // n*4 + h
  int w = threadIdx.x >> 5;     // 0..3
  long n = nh >> 2;
  int h = nh & 3;
  const float* S = softk + n * (long)Lu * 256 + h * 64;
  const float* V = vbuf + n * (long)Lu * 256 + h * 64;
  int d0 = w << 4;
  v8f acc0 = {0.f,0.f,0.f,0.f,0.f,0.f,0.f,0.f};
  v8f acc1 = acc0, acc2 = acc0, acc3 = acc0;
  for (int k0 = 0; k0 < Lu; k0 += 32) {
    v16h a = frag_a_tr_kclamp(S, 256, d0, k0, Lu);
    acc0 = wmma_f16(a, frag_b_nn_kclamp(V, 256,  0, k0, Lu), acc0);
    acc1 = wmma_f16(a, frag_b_nn_kclamp(V, 256, 16, k0, Lu), acc1);
    acc2 = wmma_f16(a, frag_b_nn_kclamp(V, 256, 32, k0, Lu), acc2);
    acc3 = wmma_f16(a, frag_b_nn_kclamp(V, 256, 48, k0, Lu), acc3);
  }
  int l = laneid();
  int ecol = l & 15;
  int rb = (l & 16) ? 8 : 0;
  float* O = kw + (long)nh * 4096;
  const float SCALE = 0.125f; // 64^-0.5
#pragma unroll
  for (int r = 0; r < 8; ++r) {
    int d = d0 + rb + r;
    O[(long)d * 64 +  0 + ecol] = acc0[r] * SCALE;
    O[(long)d * 64 + 16 + ecol] = acc1[r] * SCALE;
    O[(long)d * 64 + 32 + ecol] = acc2[r] * SCALE;
    O[(long)d * 64 + 48 + ecol] = acc3[r] * SCALE;
  }
}

// ---------------------------------------------------------------------------
// 7) out[b,h,n,e] = sum_d softq[n,d] * kw[d,e]  (per sample-head, tiled over n)
// ---------------------------------------------------------------------------
__global__ void attn_out_wmma_kernel(const float* __restrict__ q,
                                     const float* __restrict__ kw,
                                     float* __restrict__ out, int N, int Lu) {
  int LT = (Lu + 15) >> 4;
  int wid = blockIdx.x * (blockDim.x >> 5) + (threadIdx.x >> 5);
  int total = N * 4 * LT * 4;
  if (wid >= total) return;
  int et = wid & 3;
  int t1 = wid >> 2;
  int lt = t1 % LT;
  int t2 = t1 / LT;
  int h = t2 & 3;
  long n = t2 >> 2;
  const float* Qs = q + n * (long)Lu * 256 + h * 64;
  const float* Kw = kw + ((long)(n * 4 + h)) * 4096;
  v8f acc = {0.f,0.f,0.f,0.f,0.f,0.f,0.f,0.f};
#pragma unroll
  for (int k0 = 0; k0 < 64; k0 += 32) {
    v16h a = frag_a_rm_rowclamp(Qs, 256, lt << 4, Lu, k0);
    v16h b = frag_b_nn_fast(Kw, 64, et << 4, k0);
    acc = wmma_f16(a, b, acc);
  }
  int l = laneid();
  int col = l & 15;
  int rb = (l & 16) ? 8 : 0;
#pragma unroll
  for (int r = 0; r < 8; ++r) {
    int row = (lt << 4) + rb + r;
    if (row < Lu)
      out[((long)(n * Lu + row)) * 256 + h * 64 + (et << 4) + col] = acc[r];
  }
}

// ---------------------------------------------------------------------------
// 8) FFN dense conv (k=5, pad=2, 128->128) + per-sample global GN + PReLU,
//    written into cols [128,256) of the concat buffer.  One block per sample.
// ---------------------------------------------------------------------------
__global__ void dw_block_kernel(const float* __restrict__ h2, const float* __restrict__ w,
                                const float* __restrict__ bias, const float* __restrict__ gg,
                                const float* __restrict__ gb, const float* __restrict__ pr,
                                float* __restrict__ cat, int Lu) {
  __shared__ float sS[256], sSS[256];
  int n = blockIdx.x, tid = threadIdx.x;
  const float* hb = h2 + (long)n * Lu * 128;
  float* ob = cat + (long)n * Lu * 256;
  int tot = 128 * Lu;
  float s = 0.f, ss = 0.f;
  for (int idx = tid; idx < tot; idx += 256) {
    int oc = idx / Lu, l = idx % Lu;
    float acc = bias[oc];
    for (int dk = 0; dk < 5; ++dk) {
      int pos = l + dk - 2;
      if ((unsigned)pos >= (unsigned)Lu) continue;
      const float* hp = hb + (long)pos * 128;
      const float* wp = w + ((long)oc * 128) * 5 + dk;
      for (int ic = 0; ic < 128; ++ic) acc += hp[ic] * wp[ic * 5];
    }
    ob[(long)l * 256 + 128 + oc] = acc;
    s += acc; ss += acc * acc;
  }
  sS[tid] = s; sSS[tid] = ss; __syncthreads();
  for (int st = 128; st > 0; st >>= 1) {
    if (tid < st) { sS[tid] += sS[tid + st]; sSS[tid] += sSS[tid + st]; }
    __syncthreads();
  }
  float mu = sS[0] / (float)tot;
  float var = sSS[0] / (float)tot - mu * mu;
  float rinv = rsqrtf(var + 1e-5f);
  for (int idx = tid; idx < tot; idx += 256) {
    int oc = idx / Lu, l = idx % Lu;
    float v = (ob[(long)l * 256 + 128 + oc] - mu) * rinv * gg[oc] + gb[oc];
    ob[(long)l * 256 + 128 + oc] = (v >= 0.f) ? v : pr[oc] * v;
  }
}

// ---------------------------------------------------------------------------
// 9) Transposed conv (k=4, stride 1) + residual add + un-permute into 4D.
//    out[n,o,t] = b[o] + sum_{h,i: 0<=t+h-3<Lu} y3[n, t+h-3, i] * w[i,o,3-h]
// ---------------------------------------------------------------------------
__global__ void convt_out_kernel(const float* __restrict__ y3, const float* __restrict__ w,
                                 const float* __restrict__ b, const float* __restrict__ addsrc,
                                 float* __restrict__ dst, int N, int Lu, int Lin,
                                 int T, int Q, int mode) {
  long id = (long)blockIdx.x * blockDim.x + threadIdx.x;
  long total = (long)N * 64 * Lin;
  if (id >= total) return;
  int pos = (int)(id % Lin);
  long r = id / Lin;
  int o = (int)(r % 64);
  long n = r / 64;
  float acc = b[o];
  const float* yb = y3 + n * (long)Lu * 64;
  for (int h = 0; h < 4; ++h) {
    int src = pos + h - 3;
    if ((unsigned)src >= (unsigned)Lu) continue;
    const float* yp = yb + (long)src * 64;
    int koff = 3 - h;
    for (int i = 0; i < 64; ++i) acc += yp[i] * w[((long)i * 64 + o) * 4 + koff];
  }
  long out_idx;
  if (mode == 0) {
    int t = (int)(n % T); int bb = (int)(n / T);
    out_idx = (((long)bb * 64 + o) * T + t) * Q + pos;
  } else {
    int qq = (int)(n % Q); int bb = (int)(n / Q);
    out_idx = (((long)bb * 64 + o) * T + pos) * Q + qq;
  }
  dst[out_idx] = acc + addsrc[out_idx];
}

// ---------------------------------------------------------------------------
// Host-side orchestration
// ---------------------------------------------------------------------------
struct Params {
  const float *norm_g, *norm_b;
  const float *b0_cw, *b0_cb, *cw, *cb, *gn_g, *gn_b, *pr;
  const float *att_ln_g, *att_ln_b;
  const float *qk_w, *q_w, *k_w, *v_w, *out_w, *out_b;
  const float *ffw_ln_g, *ffw_ln_b;
  const float *pw1_w, *pw1_b, *pw2_w, *pw2_b;
  const float *dw_w, *dw_b, *dw_gn_g, *dw_gn_b, *dw_pr;
  const float *pw3_w, *pw3_b, *lin_w, *lin_b;
};

static Params load_params(void* const* d_in, int base) {
  Params p;
  const float** f = (const float**)&p;
  for (int i = 0; i < 32; ++i) f[i] = (const float*)d_in[base + i];
  return p;
}

static inline long cdivl(long a, long b) { return (a + b - 1) / b; }

// workspace layout (float element offsets)
static const long OFF_FF   = 0L;
static const long OFF_FEAT = OFF_FF   + 4194304L;
static const long OFF_SDBA = OFF_FEAT + 4194304L;
static const long OFF_SDBB = OFF_SDBA + 4194304L;
static const long OFF_Y    = OFF_SDBB + 4194304L;
static const long OFF_Y2   = OFF_Y    + 4194304L;
static const long OFF_Q    = OFF_Y2   + 4194304L;
static const long OFF_K    = OFF_Q    + 16777216L;
static const long OFF_V    = OFF_K    + 16777216L;
static const long OFF_KW   = OFF_V    + 16777216L;
static const long OFF_CAT  = OFF_KW   + 8388608L;
static const long OFF_H2   = OFF_CAT  + 16777216L;

static void run_pass(const float* src4d, const Params& P, float* dst4d, float* ws,
                     int mode, hipStream_t stream) {
  const int B = 2, T = 256, Q = 128;
  const int N   = (mode == 0) ? B * T : B * Q;
  const int Lin = (mode == 0) ? Q : T;
  const int Lu  = Lin - 3;           // (Lin - KS)/HS + 1 with KS=4, HS=1
  const long Ntok = (long)N * Lu;
  const int LT = (Lu + 15) >> 4;

  float* feat = ws + OFF_FEAT;
  float* sdbA = ws + OFF_SDBA;
  float* sdbB = ws + OFF_SDBB;
  float* y    = ws + OFF_Y;
  float* y2   = ws + OFF_Y2;
  float* qb   = ws + OFF_Q;
  float* kb   = ws + OFF_K;   // later reused as attention output
  float* vb   = ws + OFF_V;
  float* kwb  = ws + OFF_KW;
  float* cat  = ws + OFF_CAT;
  float* h2b  = ws + OFF_H2;
  float* yn   = sdbA;  // reuse after SDB finishes
  float* qk   = sdbB;  // reuse after SDB finishes

  // 1) ln4d + permute
  {
    long tot = (long)B * T * Q;
    ln4d_permute_kernel<<<(int)cdivl(tot, 256), 256, 0, stream>>>(
        src4d, P.norm_g, P.norm_b, feat, B, T, Q, mode);
  }
  // 2) SDB chain (4 stages; ping-pong sdbA/sdbB, final result in sdbB)
  sdb_stage_kernel<<<N, 256, 0, stream>>>(feat, Lin, nullptr, P.b0_cw, P.b0_cb,
                                          P.gn_g, P.gn_b, P.pr, sdbA, Lu, 0);
  sdb_stage_kernel<<<N, 256, 0, stream>>>(feat, Lin, sdbA, P.cw, P.cb,
                                          P.gn_g, P.gn_b, P.pr, sdbB, Lu, 1);
  sdb_stage_kernel<<<N, 256, 0, stream>>>(feat, Lin, sdbB, P.cw + 1L * 64 * 128 * 3,
                                          P.cb + 64, P.gn_g, P.gn_b, P.pr, sdbA, Lu, 2);
  sdb_stage_kernel<<<N, 256, 0, stream>>>(feat, Lin, sdbA, P.cw + 2L * 64 * 128 * 3,
                                          P.cb + 128, P.gn_g, P.gn_b, P.pr, sdbB, Lu, 3);
  // 3) (N,C,Lu) -> tokens
  ncl_to_tok_kernel<<<(int)cdivl((long)N * 64 * Lu, 256), 256, 0, stream>>>(sdbB, y, N, Lu);
  // ---- attention ----
  token_ln_kernel<<<(int)cdivl(Ntok, 256), 256, 0, stream>>>(y, P.att_ln_g, P.att_ln_b, yn, Ntok);
  qkconv_glu_kernel<<<(int)cdivl(Ntok * 64, 256), 256, 0, stream>>>(yn, P.qk_w, qk, N, Lu);
  launch_gemm(qk, P.q_w, nullptr, nullptr, qb, (int)Ntok, 256, 64, 256, 0, ACT_NONE, stream);
  launch_gemm(qk, P.k_w, nullptr, nullptr, kb, (int)Ntok, 256, 64, 256, 0, ACT_NONE, stream);
  launch_gemm(yn, P.v_w, nullptr, nullptr, vb, (int)Ntok, 256, 64, 256, 0, ACT_NONE, stream);
  softmax_col_kernel<<<(int)cdivl((long)N * 256, 256), 256, 0, stream>>>(kb, N, Lu);
  attn_kw_wmma_kernel<<<N * 4, 128, 0, stream>>>(kb, vb, kwb, Lu);
  softmax_row64_kernel<<<(int)cdivl(Ntok * 4, 256), 256, 0, stream>>>(qb, Ntok);
  {
    int total = N * 4 * LT * 4;
    attn_out_wmma_kernel<<<(total + 7) / 8, 256, 0, stream>>>(qb, kwb, kb, N, Lu);
  }
  launch_gemm(kb, P.out_w, P.out_b, y, y2, (int)Ntok, 64, 256, 64, 0, ACT_NONE, stream);
  // ---- FFN ----
  token_ln_kernel<<<(int)cdivl(Ntok, 256), 256, 0, stream>>>(y2, P.ffw_ln_g, P.ffw_ln_b, yn, Ntok);
  launch_gemm(yn, P.pw1_w, P.pw1_b, nullptr, cat, (int)Ntok, 128, 64, 256, 0, ACT_GELU, stream);
  launch_gemm(yn, P.pw2_w, P.pw2_b, nullptr, h2b, (int)Ntok, 128, 64, 128, 0, ACT_GELU, stream);
  dw_block_kernel<<<N, 256, 0, stream>>>(h2b, P.dw_w, P.dw_b, P.dw_gn_g, P.dw_gn_b, P.dw_pr,
                                         cat, Lu);
  launch_gemm(cat, P.pw3_w, P.pw3_b, y2, y, (int)Ntok, 64, 256, 64, 0, ACT_NONE, stream);
  // ---- transposed conv + residual + un-permute ----
  convt_out_kernel<<<(int)cdivl((long)N * 64 * Lin, 256), 256, 0, stream>>>(
      y, P.lin_w, P.lin_b, src4d, dst4d, N, Lu, Lin, T, Q, mode);
}

extern "C" void kernel_launch(void* const* d_in, const int* in_sizes, int n_in,
                              void* d_out, int out_size, void* d_ws, size_t ws_size,
                              hipStream_t stream) {
  (void)in_sizes; (void)n_in; (void)out_size; (void)ws_size;
  const float* x = (const float*)d_in[0];
  Params Pf = load_params(d_in, 1);
  Params Pt = load_params(d_in, 33);
  float* ws = (float*)d_ws;
  float* Ff = ws + OFF_FF;
  // freq path: Ff = process_f(x) + x
  run_pass(x, Pf, Ff, ws, 0, stream);
  // time path: out = process_t(Ff) + Ff
  run_pass(Ff, Pt, (float*)d_out, ws, 1, stream);
}
